// DeformableAttention_67834713473042
// MI455X (gfx1250) — compile-verified, matrix operand order checked by reading, not run on hardware
//
#include <hip/hip_runtime.h>
#include <hip/hip_bf16.h>

// ---------------- problem constants ----------------
#define BB   4
#define LL   9216
#define DIMV 256
#define NHH  8
#define NPP  4
#define HDD  32          // DIMV / NHH
#define SIDE 96          // sqrt(LL)
#define MTOT (BB * LL)   // 36864 rows for all GEMMs
#define KK   256         // inner dim for every GEMM here

typedef __bf16 bf16_t;
typedef __attribute__((ext_vector_type(16))) __bf16 v16bf;
typedef __attribute__((ext_vector_type(8)))  __bf16 v8bf;
typedef __attribute__((ext_vector_type(8)))  float  v8f;

// ---------------- conversion kernels ----------------
__global__ void cvt_f32_to_bf16(const float* __restrict__ in,
                                bf16_t* __restrict__ out, int n) {
    int i = blockIdx.x * blockDim.x + threadIdx.x;
    if (i < n) out[i] = (bf16_t)in[i];
}

// in: [K,N] f32 row-major  ->  out: [N,K] bf16 (transposed)
__global__ void cvt_transpose_bf16(const float* __restrict__ in,
                                   bf16_t* __restrict__ out, int K, int N) {
    int i = blockIdx.x * blockDim.x + threadIdx.x;
    if (i < K * N) {
        int k = i / N;
        int n = i % N;
        out[(size_t)n * K + k] = (bf16_t)in[i];
    }
}

// Build a 16-element bf16 WMMA fragment from two contiguous 8-element chunks.
// ISA 7.12.2 (16-bit A 16x32): lanes 0..15 hold K{0..7,16..23}, lanes 16..31
// hold K{8..15,24..31}; pointer is already offset by khalf.
__device__ __forceinline__ v16bf load_frag(const bf16_t* p) {
    v8bf lo = *(const v8bf*)(p);
    v8bf hi = *(const v8bf*)(p + 16);
    v16bf f;
#pragma unroll
    for (int i = 0; i < 8; ++i) { f[i] = lo[i]; f[i + 8] = hi[i]; }
    return f;
}

// ---------------- blocked bf16 WMMA GEMM with LDS-staged B ----------------
// C[M,N] (f32) = A[M,256] (bf16) * Bt[N,256]^T (bf16) + bias[N]
// Block = 256 threads = 8 waves. Block macro-tile: 256 rows x (NT*16) cols.
// Wave w handles rows [blockM + 32w, blockM + 32w + 31] (2 M-tiles) across
// all NT N-tiles -> 2*NT accumulators, A fragments reused NT times, B tile
// staged once in LDS and reused by all 8 waves (ds_load path).
template <int NT>
__global__ void gemm_bf16_wmma_lds(const bf16_t* __restrict__ A,
                                   const bf16_t* __restrict__ Bt,
                                   const float*  __restrict__ bias,
                                   float* __restrict__ C,
                                   int M, int N) {
    __shared__ bf16_t lbs[NT * 16 * KK];           // NT=4 -> 32 KB

    const int nBlocks = N / (NT * 16);
    const int blockM  = (blockIdx.x / nBlocks) * 256;
    const int nBase   = (blockIdx.x % nBlocks) * (NT * 16);

    // ---- cooperative fill: Bt rows [nBase, nBase+NT*16) are contiguous ----
    {
        const uint4* __restrict__ src = (const uint4*)(Bt + (size_t)nBase * KK);
        uint4* dst = (uint4*)lbs;
        const int nvec = NT * 16 * KK / 8;         // 16B vectors
#pragma unroll 2
        for (int i = threadIdx.x; i < nvec; i += 256) dst[i] = src[i];
    }
    __syncthreads();

    const int wv    = threadIdx.x >> 5;
    const int lane  = threadIdx.x & 31;
    const int khalf = (lane >> 4) * 8;             // 0 or 8
    const int l15   = lane & 15;

    const bf16_t* __restrict__ a0row = A + (size_t)(blockM + wv * 32 + l15) * KK;
    const bf16_t* __restrict__ a1row = a0row + (size_t)16 * KK;

    v8f acc[2][NT];
#pragma unroll
    for (int t = 0; t < 2; ++t)
#pragma unroll
        for (int j = 0; j < NT; ++j) acc[t][j] = (v8f){};

#pragma unroll
    for (int k0 = 0; k0 < KK; k0 += 32) {
        __builtin_prefetch(a0row + k0 + 64, 0, 1);
        __builtin_prefetch(a1row + k0 + 64, 0, 1);
        v16bf a0 = load_frag(a0row + k0 + khalf);
        v16bf a1 = load_frag(a1row + k0 + khalf);
#pragma unroll
        for (int j = 0; j < NT; ++j) {
            const bf16_t* bl = lbs + (size_t)(j * 16 + l15) * KK + k0 + khalf;
            v16bf b = load_frag(bl);               // ds_load_b128 x2
            acc[0][j] = __builtin_amdgcn_wmma_f32_16x16x32_bf16(
                false, a0, false, b, (short)0, acc[0][j], false, false);
            acc[1][j] = __builtin_amdgcn_wmma_f32_16x16x32_bf16(
                false, a1, false, b, (short)0, acc[1][j], false, false);
        }
    }

    // D layout: VGPR r -> lanes 0..15 row base+r, lanes 16..31 row base+8+r
    const int rhalf = (lane >> 4) ? 8 : 0;
#pragma unroll
    for (int t = 0; t < 2; ++t) {
        const int rowbase = blockM + wv * 32 + t * 16 + rhalf;
#pragma unroll
        for (int j = 0; j < NT; ++j) {
            const int col  = nBase + j * 16 + l15;
            const float bc = bias[col];
#pragma unroll
            for (int r = 0; r < 8; ++r) {
                C[(size_t)(rowbase + r) * N + col] = acc[t][j][r] + bc;
            }
        }
    }
}

// ---------------- deformable sampling ----------------
// vproj : [B*L, 256] f32 (head h occupies cols h*32 .. h*32+31)
// offs  : [B*L, 64]  f32 laid out (h, p, 2)
// logit : [B*L, 32]  f32 laid out (h, p)
// outp  : [B*L, 256] bf16 (pre output-projection)
// One wave per (b, q, h); lane = head-dim element -> 128B coalesced gathers.
__global__ void deform_sample(const float* __restrict__ vproj,
                              const float* __restrict__ offs,
                              const float* __restrict__ logit,
                              bf16_t* __restrict__ outp) {
    const int wave = (int)((blockIdx.x * blockDim.x + threadIdx.x) >> 5);
    const int lane = threadIdx.x & 31;
    const int h  = wave & (NHH - 1);
    const int bq = wave >> 3;            // b*L + q
    const int q  = bq % LL;
    const int b  = bq / LL;

    const int   rq = q / SIDE, cq = q % SIDE;
    const float inv = 1.0f / (float)(SIDE - 1);
    const float xr = (float)cq * inv;
    const float yr = (float)rq * inv;

    const float* __restrict__ off = offs  + (size_t)bq * (NHH * NPP * 2) + h * (NPP * 2);
    const float* __restrict__ lg  = logit + (size_t)bq * (NHH * NPP)     + h * NPP;

    float l0 = lg[0], l1 = lg[1], l2 = lg[2], l3 = lg[3];
    float mx = fmaxf(fmaxf(l0, l1), fmaxf(l2, l3));
    float e0 = __expf(l0 - mx), e1 = __expf(l1 - mx);
    float e2 = __expf(l2 - mx), e3 = __expf(l3 - mx);
    float rs = 1.0f / (e0 + e1 + e2 + e3);
    float w[NPP] = {e0 * rs, e1 * rs, e2 * rs, e3 * rs};

    const float* __restrict__ base =
        vproj + ((size_t)b * LL) * DIMV + h * HDD + lane;

    float accd = 0.0f;
#pragma unroll
    for (int p = 0; p < NPP; ++p) {
        // faithful to source: component 0 (x) -> row, component 1 (y) -> col
        float lx = xr + off[p * 2 + 0];
        float ly = yr + off[p * 2 + 1];
        lx = fminf(fmaxf(lx, 0.0f), 1.0f);
        ly = fminf(fmaxf(ly, 0.0f), 1.0f);
        float rows = lx * (float)(SIDE - 1);
        float cols = ly * (float)(SIDE - 1);
        float r0f = floorf(rows), c0f = floorf(cols);
        float fr = rows - r0f,    fc = cols - c0f;
        int r0 = (int)r0f, c0 = (int)c0f;
        int r1 = (r0 + 1 < SIDE - 1) ? r0 + 1 : SIDE - 1;
        int c1 = (c0 + 1 < SIDE - 1) ? c0 + 1 : SIDE - 1;

        float g00 = base[(size_t)(r0 * SIDE + c0) * DIMV];
        float g01 = base[(size_t)(r0 * SIDE + c1) * DIMV];
        float g10 = base[(size_t)(r1 * SIDE + c0) * DIMV];
        float g11 = base[(size_t)(r1 * SIDE + c1) * DIMV];

        float s = g00 * (1.0f - fr) * (1.0f - fc)
                + g01 * (1.0f - fr) * fc
                + g10 * fr * (1.0f - fc)
                + g11 * fr * fc;
        accd += w[p] * s;
    }
    outp[(size_t)bq * DIMV + h * HDD + lane] = (bf16_t)accd;
}

// ---------------- host-side launcher ----------------
extern "C" void kernel_launch(void* const* d_in, const int* in_sizes, int n_in,
                              void* d_out, int out_size, void* d_ws, size_t ws_size,
                              hipStream_t stream) {
    const float* query = (const float*)d_in[0];
    // d_in[1] = key : unused by the reference
    const float* value = (const float*)d_in[2];
    const float* Wv    = (const float*)d_in[3];
    const float* bv    = (const float*)d_in[4];
    const float* Woff  = (const float*)d_in[5];
    const float* boff  = (const float*)d_in[6];
    const float* Wwt   = (const float*)d_in[7];
    const float* bwt   = (const float*)d_in[8];
    const float* Wo    = (const float*)d_in[9];
    const float* bo    = (const float*)d_in[10];
    float* out = (float*)d_out;

    // -------- workspace layout (all sizes 256B-aligned) --------
    char* ws = (char*)d_ws;
    size_t o = 0;
    bf16_t* q_bf   = (bf16_t*)(ws + o); o += (size_t)MTOT * DIMV * 2;            // 18.9 MB
    bf16_t* v_bf   = (bf16_t*)(ws + o); o += (size_t)MTOT * DIMV * 2;            // 18.9 MB
    bf16_t* WvT    = (bf16_t*)(ws + o); o += (size_t)DIMV * DIMV * 2;            // 128 KB
    bf16_t* WoffT  = (bf16_t*)(ws + o); o += (size_t)(NHH * NPP * 2) * DIMV * 2; // 32 KB
    bf16_t* WwtT   = (bf16_t*)(ws + o); o += (size_t)(NHH * NPP) * DIMV * 2;     // 16 KB
    bf16_t* WoT    = (bf16_t*)(ws + o); o += (size_t)DIMV * DIMV * 2;            // 128 KB
    float*  vproj  = (float*)(ws + o);  o += (size_t)MTOT * DIMV * 4;            // 37.7 MB
    float*  offsb  = (float*)(ws + o);  o += (size_t)MTOT * (NHH * NPP * 2) * 4; // 9.4 MB
    float*  wtl    = (float*)(ws + o);  o += (size_t)MTOT * (NHH * NPP) * 4;     // 4.7 MB
    bf16_t* pre_bf = (bf16_t*)(ws + o); o += (size_t)MTOT * DIMV * 2;            // 18.9 MB
    (void)ws_size; (void)n_in; (void)in_sizes; (void)out_size;

    const int TPB = 256;

    // 1) activation conversions
    {
        int n = MTOT * DIMV;
        cvt_f32_to_bf16<<<(n + TPB - 1) / TPB, TPB, 0, stream>>>(query, q_bf, n);
        cvt_f32_to_bf16<<<(n + TPB - 1) / TPB, TPB, 0, stream>>>(value, v_bf, n);
    }
    // 2) weight transpose-conversions  ([K,N] f32 -> [N,K] bf16)
    {
        int n;
        n = DIMV * DIMV;
        cvt_transpose_bf16<<<(n + TPB - 1) / TPB, TPB, 0, stream>>>(Wv, WvT, DIMV, DIMV);
        n = DIMV * (NHH * NPP * 2);
        cvt_transpose_bf16<<<(n + TPB - 1) / TPB, TPB, 0, stream>>>(Woff, WoffT, DIMV, NHH * NPP * 2);
        n = DIMV * (NHH * NPP);
        cvt_transpose_bf16<<<(n + TPB - 1) / TPB, TPB, 0, stream>>>(Wwt, WwtT, DIMV, NHH * NPP);
        n = DIMV * DIMV;
        cvt_transpose_bf16<<<(n + TPB - 1) / TPB, TPB, 0, stream>>>(Wo, WoT, DIMV, DIMV);
    }

    // GEMM grids: block macro-tile = 256 rows x (NT*16) cols, 256 threads.
    const int mBlocks = MTOT / 256;                 // 144

    // 3) value projection: vproj = value @ Wv + bv      [36864 x 256]
    gemm_bf16_wmma_lds<4><<<mBlocks * (DIMV / 64), TPB, 0, stream>>>(
        v_bf, WvT, bv, vproj, MTOT, DIMV);

    // 4) offsets: offs = query @ Woff + boff            [36864 x 64]
    gemm_bf16_wmma_lds<4><<<mBlocks * 1, TPB, 0, stream>>>(
        q_bf, WoffT, boff, offsb, MTOT, NHH * NPP * 2);

    // 5) weight logits: wtl = query @ Wwt + bwt         [36864 x 32]
    gemm_bf16_wmma_lds<2><<<mBlocks * 1, TPB, 0, stream>>>(
        q_bf, WwtT, bwt, wtl, MTOT, NHH * NPP);

    // 6) softmax + bilinear deformable sampling -> pre_bf (bf16)
    {
        long long waves = (long long)BB * LL * NHH;   // 294912
        int blocks = (int)((waves * 32 + TPB - 1) / TPB);
        deform_sample<<<blocks, TPB, 0, stream>>>(vproj, offsb, wtl, pre_bf);
    }

    // 7) output projection: out = pre @ Wo + bo         [36864 x 256]
    gemm_bf16_wmma_lds<4><<<mBlocks * (DIMV / 64), TPB, 0, stream>>>(
        pre_bf, WoT, bo, out, MTOT, DIMV);
}
